// SelfAttention_32014686224684
// MI455X (gfx1250) — matrix operand order repaired and checked by reading, not compiled, not run on hardware
//
#include <hip/hip_runtime.h>
#include <hip/hip_bf16.h>
#include <math.h>
#include <stdint.h>

// ---------------------------------------------------------------------------
// Self-attention on MI455X (gfx1250): all matmuls via v_wmma_f32_16x16x32_bf16
// Pipeline: 3x projection GEMM (f32->bf16) -> scores GEMM (+scale+mask) ->
//           row softmax -> P@V GEMM (f32 + bf16 out) -> final GEMM (f32 out)
// Round 3: fix async-copy builtin arg types: params are v4i (vector_size 16)
//          pointers in AS1 (global src) / AS3 (LDS dst), non-const.
// ---------------------------------------------------------------------------

typedef __bf16 bf16_t;
typedef __attribute__((ext_vector_type(16))) __bf16 v16bf;
typedef __attribute__((ext_vector_type(8)))  float  v8f;
typedef int v4i_n __attribute__((vector_size(16)));   // native clang vector

#define EPI_BF16     0
#define EPI_SCORES   1
#define EPI_F32      2
#define EPI_F32_BF16 3

#if defined(__gfx1250__) && __has_builtin(__builtin_amdgcn_global_load_async_to_lds_b128)
#define HAVE_ASYNC_LDS 1
#else
#define HAVE_ASYNC_LDS 0
#endif

union FragB16 { v16bf v; uint4 q[2]; };
union Pack16  { bf16_t b[16]; uint4 q[2]; };

template<typename T> inline constexpr bool is_bf16_v = false;
template<>           inline constexpr bool is_bf16_v<bf16_t> = true;

// load 16 contiguous elements and convert (if needed) to bf16
__device__ __forceinline__ void load16(const float* s, bf16_t* d) {
  const float4* sv = (const float4*)s;
#pragma unroll
  for (int i = 0; i < 4; ++i) {
    float4 t = sv[i];
    d[4*i+0] = (bf16_t)t.x; d[4*i+1] = (bf16_t)t.y;
    d[4*i+2] = (bf16_t)t.z; d[4*i+3] = (bf16_t)t.w;
  }
}
__device__ __forceinline__ void load16(const bf16_t* s, bf16_t* d) {
  ((uint4*)d)[0] = ((const uint4*)s)[0];
  ((uint4*)d)[1] = ((const uint4*)s)[1];
}

#if HAVE_ASYNC_LDS
// 16B global -> LDS async copy (ASYNCcnt-tracked, no VGPR round trip)
__device__ __forceinline__ void async_copy16B(const bf16_t* g, bf16_t* l) {
  __builtin_amdgcn_global_load_async_to_lds_b128(
      (__attribute__((address_space(1))) v4i_n*)(uintptr_t)g,
      (__attribute__((address_space(3))) v4i_n*)(unsigned)(uintptr_t)l,
      0, 0);
}
__device__ __forceinline__ void wait_async0() {
#if __has_builtin(__builtin_amdgcn_s_wait_asynccnt)
  __builtin_amdgcn_s_wait_asynccnt(0);
#else
  asm volatile("s_wait_asynccnt 0x0" ::: "memory");
#endif
}
#endif

// padded LDS row stride in bf16 elems: 96 bytes -> keeps 16B alignment for b128
constexpr int LDSP = 48;

// C[M,N] = A[M,K] * B'[K,N]
//   B_KXN=false : B memory is [N][K] (weight layout; rows are contiguous-k)
//   B_KXN=true  : B memory is [K][N] (transposed into LDS during staging)
// Tile: 128x128 per block (256 thr = 8 waves, 2x4 wave grid, 4x2 16x16 tiles
// per wave), k-step 32, double-buffered LDS. Dims divide tiles exactly.
template<typename AT, typename BT, bool B_KXN, int EPI>
__global__ __launch_bounds__(256)
void gemm_wmma(const AT* __restrict__ A, const BT* __restrict__ Bm,
               float* __restrict__ Cf, bf16_t* __restrict__ Cb,
               const unsigned char* __restrict__ Mk,
               int K, int lda, int ldb, int ldcf, int ldcb, int ldm,
               long long a_bs, long long b_bs, long long cf_bs,
               long long cb_bs, long long m_bs, float scale)
{
  __shared__ __align__(16) bf16_t sA[2][128 * LDSP];
  __shared__ __align__(16) bf16_t sB[2][128 * LDSP];

  const int tid  = threadIdx.x;
  const int wave = tid >> 5;      // wave32
  const int lane = tid & 31;
  const int wm   = wave >> 2;     // 0..1 : 64-row band
  const int wn   = wave & 3;      // 0..3 : 32-col band
  const int lrow = lane & 15;
  const int lhi  = lane >> 4;

  const int m0 = blockIdx.y * 128;
  const int n0 = blockIdx.x * 128;
  const long long bz = blockIdx.z;
  A  += bz * a_bs;
  Bm += bz * b_bs;
  if (Cf) Cf += bz * cf_bs;
  if (Cb) Cb += bz * cb_bs;
  if (Mk) Mk += bz * m_bs;

#if HAVE_ASYNC_LDS
  constexpr bool ASY_A = is_bf16_v<AT>;
  constexpr bool ASY_B = (!B_KXN) && is_bf16_v<BT>;
#else
  constexpr bool ASY_A = false;
  constexpr bool ASY_B = false;
#endif

  const v8f zero = {0.f,0.f,0.f,0.f,0.f,0.f,0.f,0.f};
  v8f acc[4][2];
#pragma unroll
  for (int mt = 0; mt < 4; ++mt)
#pragma unroll
    for (int nt = 0; nt < 2; ++nt) acc[mt][nt] = zero;

  const int arow  = tid >> 1;     // reg-staged path: 0..127, 16-elem halves
  const int aseg  = tid & 1;
  const int qrow  = tid >> 2;     // async path: 0..63 (+64), 8-elem quarters
  const int qpart = tid & 3;

  // ---- staging helpers (issue = start loads; finish = LDS stores if needed)
  auto issueA = [&](int kc, bf16_t* dst, Pack16& t) {
    if constexpr (ASY_A) {
#if HAVE_ASYNC_LDS
#pragma unroll
      for (int h = 0; h < 2; ++h) {
        const int row = qrow + h * 64;
        async_copy16B((const bf16_t*)A + (size_t)(m0 + row) * lda + kc + qpart * 8,
                      dst + row * LDSP + qpart * 8);
      }
#endif
    } else {
      load16(A + (size_t)(m0 + arow) * lda + kc + aseg * 16, t.b);
    }
  };
  auto finishA = [&](bf16_t* dst, Pack16& t) {
    if constexpr (!ASY_A) {
      bf16_t* d = dst + arow * LDSP + aseg * 16;
      *(uint4*)d = t.q[0]; *(uint4*)(d + 8) = t.q[1];
    }
  };
  auto issueB = [&](int kc, bf16_t* dst, Pack16& t) {
    if constexpr (ASY_B) {
#if HAVE_ASYNC_LDS
#pragma unroll
      for (int h = 0; h < 2; ++h) {
        const int row = qrow + h * 64;
        async_copy16B((const bf16_t*)Bm + (size_t)(n0 + row) * ldb + kc + qpart * 8,
                      dst + row * LDSP + qpart * 8);
      }
#endif
    } else if constexpr (!B_KXN) {
      load16(Bm + (size_t)(n0 + arow) * ldb + kc + aseg * 16, t.b);
    } else {
      const int kk = tid >> 3;    // 0..31
      const int ns = tid & 7;     // 0..7 -> 16 cols each
      load16(Bm + (size_t)(kc + kk) * ldb + n0 + ns * 16, t.b);
    }
  };
  auto finishB = [&](bf16_t* dst, Pack16& t) {
    if constexpr (ASY_B) {
      // nothing: async writes land in LDS directly
    } else if constexpr (!B_KXN) {
      bf16_t* d = dst + arow * LDSP + aseg * 16;
      *(uint4*)d = t.q[0]; *(uint4*)(d + 8) = t.q[1];
    } else {
      const int kk = tid >> 3;
      const int ns = tid & 7;
#pragma unroll
      for (int i = 0; i < 16; ++i) dst[(ns * 16 + i) * LDSP + kk] = t.b[i];
    }
  };

  // ---- compute one 32-wide k-chunk from an LDS buffer pair
  auto compute = [&](const bf16_t* cA, const bf16_t* cB) {
    FragB16 af[4], bfr[2];
    // A frag: lane = row m; elems 0..7 <- K=8*hi.., elems 8..15 <- K=16+8*hi..
#pragma unroll
    for (int mt = 0; mt < 4; ++mt) {
      const bf16_t* p = cA + (wm * 64 + mt * 16 + lrow) * LDSP + 8 * lhi;
      af[mt].q[0] = *(const uint4*)p;
      af[mt].q[1] = *(const uint4*)(p + 16);
    }
    // B frag: lane = col n; elems 0..7 <- K=8*hi.., elems 8..15 <- K=16+8*hi..
#pragma unroll
    for (int nt = 0; nt < 2; ++nt) {
      const bf16_t* p = cB + (wn * 32 + nt * 16 + lrow) * LDSP + 8 * lhi;
      bfr[nt].q[0] = *(const uint4*)p;
      bfr[nt].q[1] = *(const uint4*)(p + 16);
    }
#pragma unroll
    for (int mt = 0; mt < 4; ++mt)
#pragma unroll
      for (int nt = 0; nt < 2; ++nt)
        acc[mt][nt] = __builtin_amdgcn_wmma_f32_16x16x32_bf16(
            false, af[mt].v, false, bfr[nt].v, (short)0, acc[mt][nt],
            false, false);
  };

  // ---- prologue: fill buffer 0
  {
    Pack16 ta, tb;
    issueA(0, sA[0], ta);
    issueB(0, sB[0], tb);
    finishA(sA[0], ta);
    finishB(sB[0], tb);
#if HAVE_ASYNC_LDS
    if constexpr (ASY_A || ASY_B) wait_async0();
#endif
    __syncthreads();
  }

  // ---- double-buffered main loop: stage chunk i+1 while computing chunk i
  int cur = 0;
  for (int kc = 32; kc < K; kc += 32) {
    Pack16 na, nb;
    issueA(kc, sA[cur ^ 1], na);
    issueB(kc, sB[cur ^ 1], nb);
    compute(sA[cur], sB[cur]);
    finishA(sA[cur ^ 1], na);
    finishB(sB[cur ^ 1], nb);
#if HAVE_ASYNC_LDS
    if constexpr (ASY_A || ASY_B) wait_async0();
#endif
    __syncthreads();
    cur ^= 1;
  }
  compute(sA[cur], sB[cur]);

  // ---- epilogue: D VGPR r -> row r + 8*hi, col = lane&15
#pragma unroll
  for (int mt = 0; mt < 4; ++mt) {
#pragma unroll
    for (int nt = 0; nt < 2; ++nt) {
      const int col = n0 + wn * 32 + nt * 16 + lrow;
#pragma unroll
      for (int r = 0; r < 8; ++r) {
        const int row = m0 + wm * 64 + mt * 16 + 8 * lhi + r;
        float vv = acc[mt][nt][r];
        if constexpr (EPI == EPI_SCORES) {
          vv *= scale;
          if (Mk[(size_t)row * ldm + col]) vv = -__builtin_inff();
          Cf[(size_t)row * ldcf + col] = vv;
        } else if constexpr (EPI == EPI_BF16) {
          Cb[(size_t)row * ldcb + col] = (bf16_t)vv;
        } else if constexpr (EPI == EPI_F32) {
          Cf[(size_t)row * ldcf + col] = vv;
        } else { // EPI_F32_BF16
          Cf[(size_t)row * ldcf + col] = vv;
          Cb[(size_t)row * ldcb + col] = (bf16_t)vv;
        }
      }
    }
  }
  (void)scale;
}

// One block per score row: softmax over L=2048 f32, write bf16 probabilities
// in place over the front half of the same f32 row (bf16 row stride = 2L).
__global__ __launch_bounds__(256)
void softmax_rows(float* __restrict__ S, int L)
{
  __shared__ float red[256];
  const int t = threadIdx.x;
  float*  srow = S + (size_t)blockIdx.x * L;
  bf16_t* prow = (bf16_t*)srow;

  float4 a = ((const float4*)srow)[t * 2 + 0];
  float4 b = ((const float4*)srow)[t * 2 + 1];
  float v[8] = {a.x, a.y, a.z, a.w, b.x, b.y, b.z, b.w};

  float m = v[0];
#pragma unroll
  for (int i = 1; i < 8; ++i) m = fmaxf(m, v[i]);
  red[t] = m; __syncthreads();
  for (int s = 128; s > 0; s >>= 1) {
    if (t < s) red[t] = fmaxf(red[t], red[t + s]);
    __syncthreads();
  }
  m = red[0]; __syncthreads();

  float e[8], sum = 0.f;
#pragma unroll
  for (int i = 0; i < 8; ++i) { e[i] = __expf(v[i] - m); sum += e[i]; }
  red[t] = sum; __syncthreads();
  for (int s = 128; s > 0; s >>= 1) {
    if (t < s) red[t] += red[t + s];
    __syncthreads();
  }
  const float inv = 1.0f / red[0];

  Pack16 o;
#pragma unroll
  for (int i = 0; i < 8; ++i) o.b[i] = (bf16_t)(e[i] * inv);
  *(uint4*)(prow + t * 8) = o.q[0];
}

extern "C" void kernel_launch(void* const* d_in, const int* in_sizes, int n_in,
                              void* d_out, int out_size, void* d_ws, size_t ws_size,
                              hipStream_t stream)
{
  (void)in_sizes; (void)n_in; (void)out_size; (void)ws_size;

  constexpr int       Bn = 4, L = 2048, IN = 1024, D = 512;
  constexpr long long ML = (long long)Bn * L;              // 8192 rows total

  const float* q   = (const float*)d_in[0];
  const float* k   = (const float*)d_in[1];
  const float* v   = (const float*)d_in[2];
  const unsigned char* mask = (const unsigned char*)d_in[3]; // bool8 storage
  const float* Wq  = (const float*)d_in[4];
  const float* Wk  = (const float*)d_in[5];
  const float* Wv  = (const float*)d_in[6];
  const float* Wfc = (const float*)d_in[7];

  float* xout   = (float*)d_out;                 // (B,L,1024)
  float* ctxout = xout + (size_t)ML * IN;        // (B,L,512)

  // workspace layout (~101 MB): qp/kp/vp/ctx bf16 (4x 8.4MB) + scores f32 (67MB)
  char* wsb = (char*)d_ws;
  bf16_t* qp = (bf16_t*)wsb;  wsb += (size_t)ML * D * sizeof(bf16_t);
  bf16_t* kp = (bf16_t*)wsb;  wsb += (size_t)ML * D * sizeof(bf16_t);
  bf16_t* vp = (bf16_t*)wsb;  wsb += (size_t)ML * D * sizeof(bf16_t);
  bf16_t* cb = (bf16_t*)wsb;  wsb += (size_t)ML * D * sizeof(bf16_t);
  float*  S  = (float*)wsb;   // B*L*L f32; bf16 P overlaid, row stride 2L elems

  dim3 blk(256);

  // 1) projections: (8192,1024) x (512,1024)^T -> bf16, K=1024
  gemm_wmma<float, float, false, EPI_BF16>
      <<<dim3(D/128, ML/128, 1), blk, 0, stream>>>(
      q, Wq, nullptr, qp, nullptr, IN, IN, IN, 0, D, 0, 0,0,0,0,0, 1.0f);
  gemm_wmma<float, float, false, EPI_BF16>
      <<<dim3(D/128, ML/128, 1), blk, 0, stream>>>(
      k, Wk, nullptr, kp, nullptr, IN, IN, IN, 0, D, 0, 0,0,0,0,0, 1.0f);
  gemm_wmma<float, float, false, EPI_BF16>
      <<<dim3(D/128, ML/128, 1), blk, 0, stream>>>(
      v, Wv, nullptr, vp, nullptr, IN, IN, IN, 0, D, 0, 0,0,0,0,0, 1.0f);

  // 2) scores: per batch S = qp @ kp^T / sqrt(64), mask -> -inf, f32 out
  gemm_wmma<bf16_t, bf16_t, false, EPI_SCORES>
      <<<dim3(L/128, L/128, Bn), blk, 0, stream>>>(
      qp, kp, S, nullptr, mask, D, D, D, L, 0, L,
      (long long)L * D, (long long)L * D, (long long)L * L, 0,
      (long long)L * L, 0.125f);

  // 3) row softmax -> bf16 probabilities (in place, row stride 2L bf16)
  softmax_rows<<<dim3(Bn * L), blk, 0, stream>>>(S, L);

  // 4) ctx = P @ vp per batch; vp memory is [K=key][N=feat] -> transposed B
  gemm_wmma<bf16_t, bf16_t, true, EPI_F32_BF16>
      <<<dim3(D/128, L/128, Bn), blk, 0, stream>>>(
      (const bf16_t*)S, vp, ctxout, cb, nullptr, L, 2 * L, D, D, D, 0,
      (long long)L * 2 * L, (long long)L * D, (long long)L * D,
      (long long)L * D, 0, 1.0f);

  // 5) x = ctx @ Wfc^T : (8192,512) x (1024,512)^T -> f32
  gemm_wmma<bf16_t, float, false, EPI_F32>
      <<<dim3(IN/128, ML/128, 1), blk, 0, stream>>>(
      cb, Wfc, xout, nullptr, nullptr, D, D, D, IN, 0, 0,
      0,0,0,0,0, 1.0f);
}
